// SPINN3d_53558242181789
// MI455X (gfx1250) — matrix operand order repaired and checked by reading, not compiled, not run on hardware
//
#include <hip/hip_runtime.h>
#include <hip/hip_bf16.h>

typedef __attribute__((ext_vector_type(16))) __bf16 v16bf;
typedef __attribute__((ext_vector_type(8)))  float  v8f;

#define NPTS 256     // points per axis
#define RDIM 128     // CP rank
#define ODIM 3       // output channels
#define MDIM (ODIM * RDIM)   // 384 rows of tower output
#define LDA 132      // padded LDS row stride (bf16 elems): 264B -> conflict-free lane stride

// ---------------------------------------------------------------------------
// Tower layer 1+2: h2[n][k] = tanh( sum_j tanh(c[n]*W1[j]+b1[j]) * W2[j][k] + b2[k] )
// grid = 256 blocks (one per point), 128 threads (one per feature)
// ---------------------------------------------------------------------------
__global__ __launch_bounds__(128) void tower_l12_kernel(
    const float* __restrict__ coord,
    const float* __restrict__ W1, const float* __restrict__ B1,
    const float* __restrict__ W2, const float* __restrict__ B2,
    float* __restrict__ h2out)            // [256][128]
{
  __shared__ float h1s[RDIM];
  const int n = blockIdx.x;
  const int t = threadIdx.x;
  const float c = coord[n];
  h1s[t] = tanhf(fmaf(c, W1[t], B1[t]));
  __syncthreads();
  float acc = B2[t];
#pragma unroll 8
  for (int j = 0; j < RDIM; ++j)
    acc = fmaf(h1s[j], W2[j * RDIM + t], acc);
  h2out[n * RDIM + t] = tanhf(acc);
}

// ---------------------------------------------------------------------------
// Tower layer 3 (linear) + transpose: f[m][n] = sum_k h2[n][k]*W3[k][m] + b3[m]
// grid = 384 blocks (one per output row m), 256 threads (one per point n)
// ---------------------------------------------------------------------------
__global__ __launch_bounds__(256) void tower_l3_kernel(
    const float* __restrict__ h2,         // [256][128]
    const float* __restrict__ W3,         // [128][384]
    const float* __restrict__ B3,         // [384]
    float* __restrict__ fout)             // [384][256]
{
  __shared__ float wcol[RDIM];
  const int m = blockIdx.x;
  const int t = threadIdx.x;
  if (t < RDIM) wcol[t] = W3[t * MDIM + m];
  __syncthreads();
  float acc = B3[m];
#pragma unroll 8
  for (int k = 0; k < RDIM; ++k)
    acc = fmaf(h2[t * RDIM + k], wcol[k], acc);
  fout[m * NPTS + t] = acc;
}

// ---------------------------------------------------------------------------
// CP contraction: per (o, x) block computes C[y,z] = sum_r (fx[r]*fy[r,y]) * fz[r,z]
// bf16 WMMA 16x16x32 with fp32 accumulation.
// ---------------------------------------------------------------------------
__device__ __forceinline__ unsigned long long ld_b64(const __bf16* p) {
  return *(const unsigned long long*)p;
}

union FragBF { v16bf v; unsigned long long q[4]; };

__global__ __launch_bounds__(256) void cp_contract_kernel(
    const float* __restrict__ f,          // [3][384][256] : f[(axis*384 + o*128 + r)*256 + n]
    float* __restrict__ out)              // [3][256][256][256]
{
  extern __shared__ char smem[];
  __bf16* aT    = (__bf16*)smem;                  // [256][LDA]  A^T: aT[y][r] = bf16(fx[r]*fy[r,y])
  __bf16* bT    = aT + NPTS * LDA;                // [256][LDA]  B^T: bT[z][r] = bf16(fz[r,z])
  float*  fxcol = (float*)(bT + NPTS * LDA);      // [128]

  const int x = blockIdx.x;
  const int o = blockIdx.y;
  const int t = threadIdx.x;

  if (t < RDIM) fxcol[t] = f[(size_t)(o * RDIM + t) * NPTS + x];        // axis 0
  __syncthreads();

  const float* fy  = f + (size_t)(1 * MDIM + o * RDIM) * NPTS;          // axis 1, rows of o
  const float* fzb = f + (size_t)(2 * MDIM + o * RDIM) * NPTS;          // axis 2
#pragma unroll 4
  for (int r = 0; r < RDIM; ++r) {
    aT[t * LDA + r] = (__bf16)(fxcol[r] * fy[r * NPTS + t]);            // y = t, coalesced read
    bT[t * LDA + r] = (__bf16)(fzb[r * NPTS + t]);                      // z = t
  }
  __syncthreads();

  const int lane = t & 31;
  const int wave = t >> 5;
  const int lm   = lane & 15;
  const int lhi  = lane >> 4;        // 0 for lanes 0-15, 1 for lanes 16-31

  for (int s = wave; s < 16; s += 8) {          // y-strip of 16 rows per iteration
    v8f acc[16];
#pragma unroll
    for (int zt = 0; zt < 16; ++zt) { v8f z = {0,0,0,0,0,0,0,0}; acc[zt] = z; }

    const int y = s * 16 + lm;                  // A fragment row M = lane&15
#pragma unroll
    for (int kk = 0; kk < 4; ++kk) {            // K chunks of 32
      // A 16x32 bf16 fragment: lane holds K = kk*32 + lhi*16 + {0..15}
      FragBF A;
      const __bf16* ap = aT + y * LDA + kk * 32 + lhi * 16;
      A.q[0] = ld_b64(ap);        A.q[1] = ld_b64(ap + 4);
      A.q[2] = ld_b64(ap + 8);    A.q[3] = ld_b64(ap + 12);

      const int kb = kk * 32 + lhi * 16;        // B 32x16: lane holds K = kb + {0..15}, N = lane&15
#pragma unroll
      for (int zt = 0; zt < 16; ++zt) {
        FragBF B;
        const __bf16* bp = bT + (zt * 16 + lm) * LDA + kb;
        B.q[0] = ld_b64(bp);      B.q[1] = ld_b64(bp + 4);
        B.q[2] = ld_b64(bp + 8);  B.q[3] = ld_b64(bp + 12);
        acc[zt] = __builtin_amdgcn_wmma_f32_16x16x32_bf16(
            false, A.v, false, B.v, (short)0, acc[zt], false, false);
      }
    }

    // store: C/D layout — VGPR v: M = v + lhi*8, N = lane&15
    const size_t obase = ((size_t)o * NPTS + x) * NPTS;
#pragma unroll
    for (int zt = 0; zt < 16; ++zt) {
      const int zz = zt * 16 + lm;
#pragma unroll
      for (int v = 0; v < 8; ++v) {
        const int yy = s * 16 + v + lhi * 8;
        out[((obase + (size_t)yy) << 8) + zz] = acc[zt][v];
      }
    }
  }
}

// ---------------------------------------------------------------------------
// Launch
// ---------------------------------------------------------------------------
extern "C" void kernel_launch(void* const* d_in, const int* in_sizes, int n_in,
                              void* d_out, int out_size, void* d_ws, size_t ws_size,
                              hipStream_t stream) {
  (void)in_sizes; (void)n_in; (void)out_size; (void)ws_size;

  // inputs: [0..2] = x,y,z coords; [3+6*axis .. ] = W1,b1,W2,b2,W3,b3 per axis
  const float* coords[ODIM];
  for (int a = 0; a < ODIM; ++a) coords[a] = (const float*)d_in[a];

  float* ws  = (float*)d_ws;
  float* h2  = ws;                                // [3][256][128]
  float* fwd = ws + (size_t)ODIM * NPTS * RDIM;   // [3][384][256]

  for (int a = 0; a < ODIM; ++a) {
    const float* W1 = (const float*)d_in[3 + 6 * a + 0];
    const float* B1 = (const float*)d_in[3 + 6 * a + 1];
    const float* W2 = (const float*)d_in[3 + 6 * a + 2];
    const float* B2 = (const float*)d_in[3 + 6 * a + 3];
    tower_l12_kernel<<<NPTS, RDIM, 0, stream>>>(
        coords[a], W1, B1, W2, B2, h2 + (size_t)a * NPTS * RDIM);
  }
  for (int a = 0; a < ODIM; ++a) {
    const float* W3 = (const float*)d_in[3 + 6 * a + 4];
    const float* B3 = (const float*)d_in[3 + 6 * a + 5];
    tower_l3_kernel<<<MDIM, NPTS, 0, stream>>>(
        h2 + (size_t)a * NPTS * RDIM, W3, B3, fwd + (size_t)a * MDIM * NPTS);
  }

  const size_t shmem = (size_t)2 * NPTS * LDA * sizeof(__bf16) + RDIM * sizeof(float);
  cp_contract_kernel<<<dim3(NPTS, ODIM), 256, shmem, stream>>>(fwd, (float*)d_out);
}
